// CGCNN_25555055411818
// MI455X (gfx1250) — matrix operand order repaired and checked by reading, not compiled, not run on hardware
//
#include <hip/hip_runtime.h>
#include <hip/hip_bf16.h>

// ---------------------------------------------------------------------------
// CGCNN forward for MI455X (gfx1250): WMMA f16 edge/node MLPs + atomic scatter
// M=32 supertiles per wave (each LDS B-fragment feeds 2 independent WMMAs),
// N split in two sequential halves to keep accumulators spill-free.
// ---------------------------------------------------------------------------

typedef __attribute__((ext_vector_type(16))) _Float16 v16h;
typedef __attribute__((ext_vector_type(8)))  float    v8f;

#define EPSBN 1e-5f

__device__ __forceinline__ v8f wmma16(v16h a, v16h b, v8f c) {
  // v_wmma_f32_16x16x32_f16: D = A(16x32 f16) * B(32x16 f16) + C(16x16 f32)
  return __builtin_amdgcn_wmma_f32_16x16x32_f16(false, a, false, b, (short)0, c,
                                                false, false);
}

__device__ __forceinline__ v8f vzero8() {
  v8f v;
#pragma unroll
  for (int i = 0; i < 8; ++i) v[i] = 0.f;
  return v;
}

// A fragment (16x32, f16) from an f32 row: lane holds row M=lane&15.
// halves 0..7  <- K = kb + aoff + 0..7      (aoff = 0 for lanes 0-15, 8 for 16-31)
// halves 8..15 <- K = kb + 16 + aoff + 0..7
__device__ __forceinline__ v16h a_frag_f32(const float* __restrict__ row, int kb, int aoff) {
  v16h a;
#pragma unroll
  for (int i = 0; i < 8; ++i) a[i] = (_Float16)row[kb + aoff + i];
#pragma unroll
  for (int i = 0; i < 8; ++i) a[8 + i] = (_Float16)row[kb + 16 + aoff + i];
  return a;
}

__device__ __forceinline__ v16h a_frag_h(const _Float16* __restrict__ row, int kb, int aoff) {
  v16h a;
#pragma unroll
  for (int i = 0; i < 8; ++i) a[i] = row[kb + aoff + i];
#pragma unroll
  for (int i = 0; i < 8; ++i) a[8 + i] = row[kb + 16 + aoff + i];
  return a;
}

// B fragment (32x16, f16) from transposed weights in LDS (col = N, contiguous K):
// halves 0..7 <- K = kb + boff/2 + 0..7 ; halves 8..15 <- K = kb + 16 + boff/2 + 0..7  (boff = 0 or 16)
__device__ __forceinline__ v16h b_frag_h(const _Float16* __restrict__ col, int kb, int boff) {
  v16h b;
#pragma unroll
  for (int i = 0; i < 8; ++i) b[i] = col[kb + (boff >> 1) + i];
#pragma unroll
  for (int i = 0; i < 8; ++i) b[8 + i] = col[kb + 16 + (boff >> 1) + i];
  return b;
}

__device__ __forceinline__ float softplus_f(float v) {
  return fmaxf(v, 0.f) + log1pf(__expf(-fabsf(v)));
}

// ---------------------------------------------------------------------------
// Utility kernels
// ---------------------------------------------------------------------------
__global__ void cg_zero_f32(float* __restrict__ p, long n) {
  long i = (long)blockIdx.x * blockDim.x + threadIdx.x;
  long s = (long)gridDim.x * blockDim.x;
  for (; i < n; i += s) p[i] = 0.f;
}

__global__ void cg_embed(const int* __restrict__ az, const float* __restrict__ emb,
                         float* __restrict__ x, int N_) {
  long n = (long)N_ * 64;
  long i = (long)blockIdx.x * blockDim.x + threadIdx.x;
  long s = (long)gridDim.x * blockDim.x;
  for (; i < n; i += s) {
    int r = (int)(i >> 6), c = (int)(i & 63);
    x[i] = emb[az[r] * 64 + c];
  }
}

// ---------------------------------------------------------------------------
// Edge message MLP + segment-sum scatter.  One wave per 32-edge supertile
// (two 16-row halves sharing each B-matrix fragment).
//   z = [x[dst] | x[src] | e_attr]  (K = 129, last col rank-1 in VALU)
//   m = sp(sp(z@W1+b1)@W2+b2); aggr[dst] += m
// ---------------------------------------------------------------------------
__global__ void __launch_bounds__(256, 1)
cgcnn_edge_kernel(const float* __restrict__ x,
                  const int* __restrict__ esrc,
                  const int* __restrict__ edst,
                  const float* __restrict__ eattr,
                  const float* __restrict__ W1,  // [129][128]
                  const float* __restrict__ b1,  // [128]
                  const float* __restrict__ W2,  // [128][128]
                  const float* __restrict__ b2,  // [128]
                  float* __restrict__ aggr,      // [N][128]
                  int E_, int ntiles) {
  extern __shared__ char smem_raw[];
  _Float16* W1t    = (_Float16*)smem_raw;        // [n:128][k:128] f16 transposed
  _Float16* W2t    = W1t + 128 * 128;            // [n:128][k:128]
  float*    w1last = (float*)(W2t + 128 * 128);  // W1 row 128 (edge_attr column)
  float*    b1s    = w1last + 128;
  float*    b2s    = b1s + 128;
  _Float16* mtile  = (_Float16*)(b2s + 128);     // [8 waves][32][128]

  for (int i = threadIdx.x; i < 128 * 128; i += blockDim.x) {
    int k = i >> 7, n = i & 127;
    W1t[n * 128 + k] = (_Float16)W1[i];
    W2t[n * 128 + k] = (_Float16)W2[i];
  }
  for (int i = threadIdx.x; i < 128; i += blockDim.x) {
    w1last[i] = W1[128 * 128 + i];
    b1s[i] = b1[i];
    b2s[i] = b2[i];
  }
  __syncthreads();

  const int lane  = threadIdx.x & 31;
  const int wv    = threadIdx.x >> 5;
  const int n0    = lane & 15;
  const int hi    = (lane >> 4) & 1;
  const int aoff  = hi ? 8 : 0;
  const int boff  = hi ? 16 : 0;
  const int mbase = hi ? 8 : 0;
  _Float16* msm = mtile + wv * (32 * 128);
  const int wstep = gridDim.x * (blockDim.x >> 5);

  for (int tile = blockIdx.x * (blockDim.x >> 5) + wv; tile < ntiles; tile += wstep) {
    const int e0 = tile * 32;
    int eA = e0 + n0;      if (eA >= E_) eA = E_ - 1;   // clamp: keep EXEC full
    int eB = e0 + 16 + n0; if (eB >= E_) eB = E_ - 1;
    const float* xdA = x + (size_t)edst[eA] * 64;
    const float* xsA = x + (size_t)esrc[eA] * 64;
    const float* xdB = x + (size_t)edst[eB] * 64;
    const float* xsB = x + (size_t)esrc[eB] * 64;

    float eavA[8], eavB[8];
#pragma unroll
    for (int r = 0; r < 8; ++r) {
      int erA = e0 + mbase + r;      if (erA >= E_) erA = E_ - 1;
      int erB = e0 + 16 + mbase + r; if (erB >= E_) erB = E_ - 1;
      eavA[r] = eattr[erA];
      eavB[r] = eattr[erB];
    }

    // ---- GEMM1: (32 x 128) = z_dense @ W1[0:128], two sequential n-halves ----
#pragma unroll 1
    for (int h2 = 0; h2 < 2; ++h2) {
      v8f accA[4], accB[4];
#pragma unroll
      for (int nt = 0; nt < 4; ++nt) { accA[nt] = vzero8(); accB[nt] = vzero8(); }
#pragma unroll
      for (int kc = 0; kc < 4; ++kc) {
        const float* apA = (kc < 2) ? xdA : xsA;        // K 0..63 = dst, 64..127 = src
        const float* apB = (kc < 2) ? xdB : xsB;
        v16h a0 = a_frag_f32(apA, (kc & 1) * 32, aoff);
        v16h a1 = a_frag_f32(apB, (kc & 1) * 32, aoff);
#pragma unroll
        for (int nt = 0; nt < 4; ++nt) {
          v16h b = b_frag_h(W1t + ((h2 * 4 + nt) * 16 + n0) * 128, kc * 32, boff);
          accA[nt] = wmma16(a0, b, accA[nt]);
          accB[nt] = wmma16(a1, b, accB[nt]);
        }
      }
      // bias + edge_attr rank-1 + softplus -> per-wave LDS tile (f16)
#pragma unroll
      for (int nt = 0; nt < 4; ++nt) {
        int n = (h2 * 4 + nt) * 16 + n0;
        float bb = b1s[n], wl = w1last[n];
#pragma unroll
        for (int r = 0; r < 8; ++r) {
          msm[(mbase + r) * 128 + n] =
              (_Float16)softplus_f(accA[nt][r] + bb + eavA[r] * wl);
          msm[(16 + mbase + r) * 128 + n] =
              (_Float16)softplus_f(accB[nt][r] + bb + eavB[r] * wl);
        }
      }
    }
    // per-wave LDS ops are in-order: GEMM2 loads below see the stores above.

    int drowA[8], drowB[8]; bool valA[8], valB[8];
#pragma unroll
    for (int r = 0; r < 8; ++r) {
      int erA = e0 + mbase + r;
      int erB = e0 + 16 + mbase + r;
      valA[r] = (erA < E_);
      valB[r] = (erB < E_);
      if (erA >= E_) erA = E_ - 1;
      if (erB >= E_) erB = E_ - 1;
      drowA[r] = edst[erA];
      drowB[r] = edst[erB];
    }

    // ---- GEMM2: (32 x 128) @ W2, two sequential n-halves + scatter ----
#pragma unroll 1
    for (int h2 = 0; h2 < 2; ++h2) {
      v8f acc2A[4], acc2B[4];
#pragma unroll
      for (int nt = 0; nt < 4; ++nt) { acc2A[nt] = vzero8(); acc2B[nt] = vzero8(); }
#pragma unroll
      for (int kc = 0; kc < 4; ++kc) {
        v16h a0 = a_frag_h(msm + n0 * 128, kc * 32, aoff);
        v16h a1 = a_frag_h(msm + (16 + n0) * 128, kc * 32, aoff);
#pragma unroll
        for (int nt = 0; nt < 4; ++nt) {
          v16h b = b_frag_h(W2t + ((h2 * 4 + nt) * 16 + n0) * 128, kc * 32, boff);
          acc2A[nt] = wmma16(a0, b, acc2A[nt]);
          acc2B[nt] = wmma16(a1, b, acc2B[nt]);
        }
      }
      // softplus + scatter-add (segment_sum over dst)
#pragma unroll
      for (int nt = 0; nt < 4; ++nt) {
        int n = (h2 * 4 + nt) * 16 + n0;
        float bb = b2s[n];
#pragma unroll
        for (int r = 0; r < 8; ++r) {
          float vA = softplus_f(acc2A[nt][r] + bb);
          float vB = softplus_f(acc2B[nt][r] + bb);
          if (valA[r]) atomicAdd(aggr + (size_t)drowA[r] * 128 + n, vA);
          if (valB[r]) atomicAdd(aggr + (size_t)drowB[r] * 128 + n, vB);
        }
      }
    }
  }
}

// ---------------------------------------------------------------------------
// Node update MLP: x = sp([x|aggr]@W1+b1)@W2 + b2 + x   (in-place rows)
// One wave per 32-node supertile, same B-fragment reuse + n-half scheme.
// ---------------------------------------------------------------------------
__global__ void __launch_bounds__(256, 1)
cgcnn_node_kernel(float* __restrict__ x,            // [N][64]
                  const float* __restrict__ aggr,   // [N][128]
                  const float* __restrict__ W1,     // [192][128]
                  const float* __restrict__ b1,     // [128]
                  const float* __restrict__ W2,     // [128][64]
                  const float* __restrict__ b2,     // [64]
                  int N_, int ntiles) {
  extern __shared__ char smem_raw[];
  _Float16* W1t   = (_Float16*)smem_raw;        // [n:128][k:192]
  _Float16* W2t   = W1t + 128 * 192;            // [n:64][k:128]
  float*    b1s   = (float*)(W2t + 64 * 128);
  float*    b2s   = b1s + 128;
  _Float16* mtile = (_Float16*)(b2s + 64);      // [8 waves][32][128]

  for (int i = threadIdx.x; i < 192 * 128; i += blockDim.x) {
    int k = i >> 7, n = i & 127;
    W1t[n * 192 + k] = (_Float16)W1[i];
  }
  for (int i = threadIdx.x; i < 128 * 64; i += blockDim.x) {
    int k = i >> 6, n = i & 63;
    W2t[n * 128 + k] = (_Float16)W2[i];
  }
  for (int i = threadIdx.x; i < 128; i += blockDim.x) b1s[i] = b1[i];
  for (int i = threadIdx.x; i < 64; i += blockDim.x)  b2s[i] = b2[i];
  __syncthreads();

  const int lane  = threadIdx.x & 31;
  const int wv    = threadIdx.x >> 5;
  const int n0    = lane & 15;
  const int hi    = (lane >> 4) & 1;
  const int aoff  = hi ? 8 : 0;
  const int boff  = hi ? 16 : 0;
  const int mbase = hi ? 8 : 0;
  _Float16* msm = mtile + wv * (32 * 128);
  const int wstep = gridDim.x * (blockDim.x >> 5);

  for (int tile = blockIdx.x * (blockDim.x >> 5) + wv; tile < ntiles; tile += wstep) {
    const int r0 = tile * 32;
    int rA = r0 + n0;      if (rA >= N_) rA = N_ - 1;
    int rB = r0 + 16 + n0; if (rB >= N_) rB = N_ - 1;
    const float* xrA = x + (size_t)rA * 64;
    const float* grA = aggr + (size_t)rA * 128;
    const float* xrB = x + (size_t)rB * 64;
    const float* grB = aggr + (size_t)rB * 128;

    // ---- GEMM1: h(32x192) @ W1 -> 32x128, two sequential n-halves ----
#pragma unroll 1
    for (int h2 = 0; h2 < 2; ++h2) {
      v8f accA[4], accB[4];
#pragma unroll
      for (int nt = 0; nt < 4; ++nt) { accA[nt] = vzero8(); accB[nt] = vzero8(); }
#pragma unroll
      for (int kc = 0; kc < 6; ++kc) {
        v16h a0 = (kc < 2) ? a_frag_f32(xrA, kc * 32, aoff)
                           : a_frag_f32(grA, (kc - 2) * 32, aoff);
        v16h a1 = (kc < 2) ? a_frag_f32(xrB, kc * 32, aoff)
                           : a_frag_f32(grB, (kc - 2) * 32, aoff);
#pragma unroll
        for (int nt = 0; nt < 4; ++nt) {
          v16h b = b_frag_h(W1t + ((h2 * 4 + nt) * 16 + n0) * 192, kc * 32, boff);
          accA[nt] = wmma16(a0, b, accA[nt]);
          accB[nt] = wmma16(a1, b, accB[nt]);
        }
      }
#pragma unroll
      for (int nt = 0; nt < 4; ++nt) {
        int n = (h2 * 4 + nt) * 16 + n0;
        float bb = b1s[n];
#pragma unroll
        for (int r = 0; r < 8; ++r) {
          msm[(mbase + r) * 128 + n]      = (_Float16)softplus_f(accA[nt][r] + bb);
          msm[(16 + mbase + r) * 128 + n] = (_Float16)softplus_f(accB[nt][r] + bb);
        }
      }
    }

    // ---- GEMM2: (32x128) @ W2 -> 32x64 (4 n-tiles: fits in one pass) ----
    v8f acc2A[4], acc2B[4];
#pragma unroll
    for (int nt = 0; nt < 4; ++nt) { acc2A[nt] = vzero8(); acc2B[nt] = vzero8(); }
#pragma unroll
    for (int kc = 0; kc < 4; ++kc) {
      v16h a0 = a_frag_h(msm + n0 * 128, kc * 32, aoff);
      v16h a1 = a_frag_h(msm + (16 + n0) * 128, kc * 32, aoff);
#pragma unroll
      for (int nt = 0; nt < 4; ++nt) {
        v16h b = b_frag_h(W2t + (nt * 16 + n0) * 128, kc * 32, boff);
        acc2A[nt] = wmma16(a0, b, acc2A[nt]);
        acc2B[nt] = wmma16(a1, b, acc2B[nt]);
      }
    }

    // residual epilogue (rows disjoint per wave -> safe in-place)
#pragma unroll
    for (int nt = 0; nt < 4; ++nt) {
      int n = nt * 16 + n0;
      float bb = b2s[n];
#pragma unroll
      for (int r = 0; r < 8; ++r) {
        int rowA = r0 + mbase + r;
        int rowB = r0 + 16 + mbase + r;
        if (rowA < N_) {
          size_t ix = (size_t)rowA * 64 + n;
          x[ix] = acc2A[nt][r] + bb + x[ix];
        }
        if (rowB < N_) {
          size_t ix = (size_t)rowB * 64 + n;
          x[ix] = acc2B[nt][r] + bb + x[ix];
        }
      }
    }
  }
}

// ---------------------------------------------------------------------------
// BatchNorm (training-mode batch stats over N rows, 64 cols)
// ---------------------------------------------------------------------------
__global__ void cg_bn_stats(const float* __restrict__ x, float* __restrict__ stats,
                            int N_) {
  // stats[0:64] = colsum, stats[64:128] = colsumsq
  int c = threadIdx.x & 63;
  int rpb = blockDim.x >> 6;
  float s = 0.f, s2 = 0.f;
  for (int r = blockIdx.x * rpb + (threadIdx.x >> 6); r < N_; r += gridDim.x * rpb) {
    float v = x[(size_t)r * 64 + c];
    s += v; s2 += v * v;
  }
  atomicAdd(&stats[c], s);
  atomicAdd(&stats[64 + c], s2);
}

__global__ void cg_bn_apply(float* __restrict__ x, const float* __restrict__ stats,
                            const float* __restrict__ gamma,
                            const float* __restrict__ beta, int N_) {
  long n = (long)N_ * 64;
  long i = (long)blockIdx.x * blockDim.x + threadIdx.x;
  long s = (long)gridDim.x * blockDim.x;
  float invN = 1.f / (float)N_;
  for (; i < n; i += s) {
    int c = (int)(i & 63);
    float mu  = stats[c] * invN;
    float var = stats[64 + c] * invN - mu * mu;
    x[i] = gamma[c] * (x[i] - mu) * rsqrtf(var + EPSBN) + beta[c];
  }
}

// ---------------------------------------------------------------------------
// Global mean pool + output MLP
// ---------------------------------------------------------------------------
__global__ void cg_pool_sum(const float* __restrict__ x, const int* __restrict__ batch,
                            float* __restrict__ gsum, float* __restrict__ gcnt, int N_) {
  long n = (long)N_ * 64;
  long i = (long)blockIdx.x * blockDim.x + threadIdx.x;
  long s = (long)gridDim.x * blockDim.x;
  for (; i < n; i += s) {
    int r = (int)(i >> 6), c = (int)(i & 63);
    int g = batch[r];
    atomicAdd(&gsum[(size_t)g * 64 + c], x[i]);
    if (c == 0) atomicAdd(&gcnt[g], 1.f);
  }
}

__global__ void cg_out_mlp(const float* __restrict__ gsum, const float* __restrict__ gcnt,
                           const float* __restrict__ W1,   // [64][128]
                           const float* __restrict__ b1,   // [128]
                           const float* __restrict__ W2,   // [128][1]
                           const float* __restrict__ b2,   // [1]
                           float* __restrict__ out, int G_) {
  int g = blockIdx.x * blockDim.x + threadIdx.x;
  if (g >= G_) return;
  float inv = 1.f / fmaxf(gcnt[g], 1.f);
  float pooled[64];
#pragma unroll
  for (int d = 0; d < 64; ++d) pooled[d] = gsum[(size_t)g * 64 + d] * inv;
  float acc = b2[0];
  for (int h = 0; h < 128; ++h) {
    float v = b1[h];
#pragma unroll
    for (int d = 0; d < 64; ++d) v += pooled[d] * W1[d * 128 + h];
    acc += softplus_f(v) * W2[h];
  }
  out[g] = acc;
}

// ---------------------------------------------------------------------------
// Host launcher
// ---------------------------------------------------------------------------
extern "C" void kernel_launch(void* const* d_in, const int* in_sizes, int n_in,
                              void* d_out, int out_size, void* d_ws, size_t ws_size,
                              hipStream_t stream) {
  const int*   atom_z = (const int*)d_in[0];
  const int*   eidx   = (const int*)d_in[1];
  const float* eattr  = (const float*)d_in[2];
  const int*   batch  = (const int*)d_in[3];
  const float* emb    = (const float*)d_in[4];
  const float* eW1    = (const float*)d_in[5];
  const float* eb1    = (const float*)d_in[6];
  const float* eW2    = (const float*)d_in[7];
  const float* eb2    = (const float*)d_in[8];
  const float* nW1    = (const float*)d_in[9];
  const float* nb1    = (const float*)d_in[10];
  const float* nW2    = (const float*)d_in[11];
  const float* nb2    = (const float*)d_in[12];
  const float* gam    = (const float*)d_in[13];
  const float* bet    = (const float*)d_in[14];
  const float* oW1    = (const float*)d_in[15];
  const float* ob1    = (const float*)d_in[16];
  const float* oW2    = (const float*)d_in[17];
  const float* ob2    = (const float*)d_in[18];

  const int N_ = in_sizes[0];
  const int E_ = in_sizes[2];      // edge_attr has E*ED (ED==1) elements
  const int G_ = out_size;         // output [G,1]
  const int L_ = 3;

  const int* esrc = eidx;          // edge_index[0] = src (x_j)
  const int* edst = eidx + E_;     // edge_index[1] = dst (x_i, aggregation target)

  float* ws    = (float*)d_ws;
  float* x     = ws;                          // N*64
  float* aggr  = x + (size_t)N_ * 64;         // N*128
  float* stats = aggr + (size_t)N_ * 128;     // 128 (colsum | colsumsq)
  float* gsum  = stats + 128;                 // G*64
  float* gcnt  = gsum + (size_t)G_ * 64;      // G

  const int etiles = (E_ + 31) / 32;          // 32-edge supertiles
  const int ntiles = (N_ + 31) / 32;          // 32-node supertiles
  constexpr int EDGE_SMEM = 128 * 128 * 2 * 2 + 3 * 128 * 4
                          + 8 * 32 * 128 * 2;                  // 132608 B
  constexpr int NODE_SMEM = 128 * 192 * 2 + 64 * 128 * 2 + (128 + 64) * 4
                          + 8 * 32 * 128 * 2;                  // 131840 B

  // x = emb[atom_z]
  cg_embed<<<1024, 256, 0, stream>>>(atom_z, emb, x, N_);

  for (int l = 0; l < L_; ++l) {
    const float* W1e = eW1 + (size_t)l * 129 * 128;
    const float* b1e = eb1 + (size_t)l * 128;
    const float* W2e = eW2 + (size_t)l * 128 * 128;
    const float* b2e = eb2 + (size_t)l * 128;
    const float* W1n = nW1 + (size_t)l * 192 * 128;
    const float* b1n = nb1 + (size_t)l * 128;
    const float* W2n = nW2 + (size_t)l * 128 * 64;
    const float* b2n = nb2 + (size_t)l * 64;

    cg_zero_f32<<<2048, 256, 0, stream>>>(aggr, (long)N_ * 128);
    cgcnn_edge_kernel<<<2048, 256, EDGE_SMEM, stream>>>(
        x, esrc, edst, eattr, W1e, b1e, W2e, b2e, aggr, E_, etiles);
    cgcnn_node_kernel<<<512, 256, NODE_SMEM, stream>>>(
        x, aggr, W1n, b1n, W2n, b2n, N_, ntiles);

    cg_zero_f32<<<1, 128, 0, stream>>>(stats, 128);
    cg_bn_stats<<<1024, 256, 0, stream>>>(x, stats, N_);
    cg_bn_apply<<<1024, 256, 0, stream>>>(x, stats, gam + (size_t)l * 64,
                                          bet + (size_t)l * 64, N_);
  }

  cg_zero_f32<<<16, 256, 0, stream>>>(gsum, (long)G_ * 64 + G_);
  cg_pool_sum<<<1024, 256, 0, stream>>>(x, batch, gsum, gcnt, N_);
  cg_out_mlp<<<(G_ + 127) / 128, 128, 0, stream>>>(gsum, gcnt, oW1, ob1, oW2, ob2,
                                                   (float*)d_out, G_);
}
